// AttentionV4_83829171683763
// MI455X (gfx1250) — matrix operand order repaired and checked by reading, not compiled, not run on hardware
//
#include <hip/hip_runtime.h>
#include <math.h>

#define CDIM   48
#define CO     144
#define NPIX   65536
#define OHW    66
#define NT     4356      // 66*66 tokens
#define NTP    4480      // padded: 35*128, multiple of 32/16/128
#define NH     8
#define CH     96        // channels per head
#define EPSN   1e-12f

typedef __attribute__((ext_vector_type(16))) _Float16 v16h;
typedef __attribute__((ext_vector_type(8)))  _Float16 v8h;
typedef __attribute__((ext_vector_type(8)))  float    v8f;
typedef __attribute__((ext_vector_type(4)))  int      v4i_t;

#if defined(__HIP_DEVICE_COMPILE__) && __has_builtin(__builtin_amdgcn_global_load_async_to_lds_b128)
#define USE_ASYNC_LDS 1
#else
#define USE_ASYNC_LDS 0
#endif

// ---------- WMMA fragment helpers (16x16x32 f16, wave32, A/B in contiguous k = 16*(lane>>4) + e) ----------

// A matrix 16x32, row-major source (global or LDS) with leading dim lda.
// lanes 0-15: K in {0..15}; lanes 16-31: K in {16..31}
__device__ __forceinline__ v16h load_a_f16(const _Float16* A, int lda, int row0, int k0, int kmax) {
  const int lane = threadIdx.x & 31;
  const int m = lane & 15, half = lane >> 4;
  const _Float16* rowp = A + (size_t)(row0 + m) * (size_t)lda;
  v16h a;
#pragma unroll
  for (int e = 0; e < 16; ++e) {
    const int v = e >> 1;
    const int k = half * 16 + e;
    const int kk = k0 + k;
    a[e] = (kk < kmax) ? rowp[kk] : (_Float16)0.f;
  }
  return a;
}

// Same but converts from an f32 source (small weight matrices).
__device__ __forceinline__ v16h load_a_f32(const float* A, int lda, int row0, int k0, int kmax) {
  const int lane = threadIdx.x & 31;
  const int m = lane & 15, half = lane >> 4;
  const float* rowp = A + (size_t)(row0 + m) * (size_t)lda;
  v16h a;
#pragma unroll
  for (int e = 0; e < 16; ++e) {
    const int v = e >> 1;
    const int k = half * 16 + e;
    const int kk = k0 + k;
    a[e] = (kk < kmax) ? (_Float16)rowp[kk] : (_Float16)0.f;
  }
  return a;
}

// B matrix 32x16 (KxN), source stored N-major (Bt[n][k], ld=ldb):
// each lane loads 16 contiguous halves; lanes 0-15 -> K=0..15, lanes 16-31 -> K=16..31.
__device__ __forceinline__ v16h load_b_f16(const _Float16* Bt, int ldb, int col0, int k0) {
  const int lane = threadIdx.x & 31;
  const int n = lane & 15, half = lane >> 4;
  const _Float16* p = Bt + (size_t)(col0 + n) * (size_t)ldb + k0 + half * 16;
  v16h b;
#pragma unroll
  for (int e = 0; e < 16; ++e) b[e] = p[e];
  return b;
}

// C/D 16x16 f32: VGPR r -> row r + 8*(lane>>4), col = lane&15
__device__ __forceinline__ void store_d(float* C, int ldc, int row0, int col0, v8f d) {
  const int lane = threadIdx.x & 31;
  const int n = lane & 15, half = lane >> 4;
#pragma unroll
  for (int r = 0; r < 8; ++r)
    C[(size_t)(row0 + r + 8 * half) * (size_t)ldc + col0 + n] = d[r];
}

__device__ __forceinline__ v8f wmma16(v16h a, v16h b, v8f c) {
  return __builtin_amdgcn_wmma_f32_16x16x32_f16(false, a, false, b, (short)0, c, false, false);
}

// 16-byte global->LDS copy: CDNA5 async path (ASYNCcnt-tracked) if exposed, else via VGPRs.
__device__ __forceinline__ void cp16B(_Float16* dst_lds, const _Float16* src_glob) {
#if USE_ASYNC_LDS
  v4i_t* sg = (v4i_t*)const_cast<_Float16*>(src_glob);   // generic reinterpret
  v4i_t* dl = (v4i_t*)dst_lds;                           // generic reinterpret
  __builtin_amdgcn_global_load_async_to_lds_b128(
      (__attribute__((address_space(1))) v4i_t*)sg,
      (__attribute__((address_space(3))) v4i_t*)dl, 0, 0);
#else
  *(v8h*)dst_lds = *(const v8h*)src_glob;
#endif
}

// Wait until at most N async-to-LDS instructions from this wave remain in flight.
// Async loads complete in order, so waiting to <=N retires everything older.
template <int N>
__device__ __forceinline__ void wait_async() {
#if USE_ASYNC_LDS && __has_builtin(__builtin_amdgcn_s_wait_asynccnt)
  __builtin_amdgcn_s_wait_asynccnt(N);
#endif
}

// ---------- stage 0: x (48x65536 f32) -> xT (65536x64 f16, zero-padded K) ----------
__global__ void attnv4_x_to_xt(const float* __restrict__ x, _Float16* __restrict__ xT) {
  const int p = blockIdx.x * blockDim.x + threadIdx.x;
#pragma unroll
  for (int c = 0; c < 64; ++c)
    xT[(size_t)p * 64 + c] = (c < CDIM) ? (_Float16)x[(size_t)c * NPIX + p] : (_Float16)0.f;
}

// ---------- stage 1: qkv = w_qkv(144x48) * x -> (144x65536) f32 ----------
__global__ __launch_bounds__(256) void attnv4_gemm_qkv(const float* __restrict__ wq,
                                                       const _Float16* __restrict__ xT,
                                                       float* __restrict__ qkv) {
  const int wid = blockIdx.x * 8 + (threadIdx.x >> 5);
  const int mt = wid % 9, nt = wid / 9;
  v8f c = {};
#pragma unroll
  for (int k0 = 0; k0 < 64; k0 += 32) {
    v16h a = load_a_f32(wq, CDIM, mt * 16, k0, CDIM);
    v16h b = load_b_f16(xT, 64, nt * 16, k0);
    c = wmma16(a, b, c);
  }
  store_d(qkv, NPIX, mt * 16, nt * 16, c);
}

// ---------- stage 2: depthwise 3x3 ----------
__global__ void attnv4_dwconv(const float* __restrict__ in, const float* __restrict__ wdw,
                              float* __restrict__ out) {
  const int idx = blockIdx.x * blockDim.x + threadIdx.x;   // over CO*NPIX
  const int o = idx >> 16, p = idx & 65535;
  const int y = p >> 8, x = p & 255;
  const float* wp = wdw + o * 9;
  const float* ip = in + (size_t)o * NPIX;
  float s = 0.f;
#pragma unroll
  for (int ky = 0; ky < 3; ++ky)
#pragma unroll
    for (int kx = 0; kx < 3; ++kx) {
      const int yy = y + ky - 1, xx = x + kx - 1;
      if (yy >= 0 && yy < 256 && xx >= 0 && xx < 256)
        s += wp[ky * 3 + kx] * ip[yy * 256 + xx];
    }
  out[idx] = s;
}

__device__ __forceinline__ float unfold_gather(const float* qkv, int part, int h, int ty, int tx, int j) {
  const int cp = j >> 4, r = j & 15, kh = r >> 2, kw = r & 3;
  const int py = ty * 4 + kh - 4, px = tx * 4 + kw - 4;
  if (py < 0 || py >= 256 || px < 0 || px >= 256) return 0.f;
  return qkv[(size_t)(part * CDIM + h * 6 + cp) * NPIX + py * 256 + px];
}

// ---------- stage 3: unfold + head split + L2 norm ----------
// qT,kT: [h][token(NTP)][96] f16 (token-major);  vC: [h][96][token(NTP)] f16
__global__ void attnv4_make_heads(const float* __restrict__ qkv, _Float16* __restrict__ qT,
                                  _Float16* __restrict__ kT, _Float16* __restrict__ vC) {
  const int idx = blockIdx.x * blockDim.x + threadIdx.x;
  if (idx >= NH * NTP) return;
  const int h = idx / NTP, t = idx % NTP;
  _Float16* qrow = qT + (size_t)(h * NTP + t) * CH;
  _Float16* krow = kT + (size_t)(h * NTP + t) * CH;
  if (t >= NT) {  // zero-pad tokens
    for (int j = 0; j < CH; ++j) {
      qrow[j] = (_Float16)0.f; krow[j] = (_Float16)0.f;
      vC[(size_t)(h * CH + j) * NTP + t] = (_Float16)0.f;
    }
    return;
  }
  const int ty = t / OHW, tx = t % OHW;
  float qss = 0.f, kss = 0.f;
  for (int j = 0; j < CH; ++j) {
    const float qv = unfold_gather(qkv, 0, h, ty, tx, j);
    const float kv = unfold_gather(qkv, 1, h, ty, tx, j);
    qss += qv * qv; kss += kv * kv;
  }
  const float qi = 1.f / fmaxf(sqrtf(qss), EPSN);
  const float ki = 1.f / fmaxf(sqrtf(kss), EPSN);
  for (int j = 0; j < CH; ++j) {
    const float qv = unfold_gather(qkv, 0, h, ty, tx, j);
    const float kv = unfold_gather(qkv, 1, h, ty, tx, j);
    const float vv = unfold_gather(qkv, 2, h, ty, tx, j);
    qrow[j] = (_Float16)(qv * qi);
    krow[j] = (_Float16)(kv * ki);
    vC[(size_t)(h * CH + j) * NTP + t] = (_Float16)vv;
  }
}

// ---------- stage 4: rowsum[n] = sum_m exp(S[n,m]*temp)   (|S|<=temp: no max needed) ----------
// Block = 4 waves / 128 rows; double-buffered async key chunks in LDS.
__global__ __launch_bounds__(128) void attnv4_rowsum(const _Float16* __restrict__ qT,
                                                     const _Float16* __restrict__ kT,
                                                     const float* __restrict__ temp,
                                                     float* __restrict__ rowsum) {
  __shared__ _Float16 sK[2][32 * CH];              // ping-pong, 2 x 6 KB
  const int h = blockIdx.y;
  const int nblock = blockIdx.x * 128;
  const int wave = threadIdx.x >> 5, lane = threadIdx.x & 31;
  const _Float16* q = qT + (size_t)h * NTP * CH;
  const _Float16* k = kT + (size_t)h * NTP * CH;
  const float th = temp[h];
  const int nw = nblock + wave * 32;

  v16h aq[2][3];                                   // this wave's 32 query rows (hoisted)
#pragma unroll
  for (int which = 0; which < 2; ++which)
#pragma unroll
    for (int kk = 0; kk < 3; ++kk)
      aq[which][kk] = load_a_f16(q, CH, nw + which * 16, kk * 32, CH);

  float acc[2][8];
#pragma unroll
  for (int w2 = 0; w2 < 2; ++w2)
#pragma unroll
    for (int r = 0; r < 8; ++r) acc[w2][r] = 0.f;

  auto stageK = [&](int m0, int buf) {             // 3 async instructions per wave
    const _Float16* src = k + (size_t)m0 * CH;     // contiguous 6144 B
    for (int g = threadIdx.x; g < 384; g += 128) cp16B(&sK[buf][g * 8], src + g * 8);
  };

  const int NIT = NTP / 32;                        // 140
  stageK(0, 0);
  for (int it = 0; it < NIT; ++it) {
    const int cur = it & 1, nxt = cur ^ 1;
    const int mnext = (it + 1 < NIT) ? (it + 1) * 32 : it * 32;  // clamp keeps wait const
    stageK(mnext, nxt);
    wait_async<3>();                               // oldest batch (chunk `it`) has landed
    __syncthreads();
    const int m0 = it * 32;
#pragma unroll
    for (int mi = 0; mi < 2; ++mi) {
      v16h bk[3];
#pragma unroll
      for (int kk = 0; kk < 3; ++kk) bk[kk] = load_b_f16(sK[cur], CH, mi * 16, kk * 32);
      const int m = m0 + mi * 16 + (lane & 15);
      const float mask = (m < NT) ? 1.f : 0.f;     // exclude padded key tokens
#pragma unroll
      for (int which = 0; which < 2; ++which) {
        v8f s = {};
#pragma unroll
        for (int kk = 0; kk < 3; ++kk) s = wmma16(aq[which][kk], bk[kk], s);
#pragma unroll
        for (int r = 0; r < 8; ++r) acc[which][r] += mask * __expf(s[r] * th);
      }
    }
    __syncthreads();                               // everyone done with buf `cur`
  }
#pragma unroll
  for (int which = 0; which < 2; ++which)
#pragma unroll
    for (int r = 0; r < 8; ++r) {                  // reduce across the 16 column-lanes
      float v = acc[which][r];
      v += __shfl_xor(v, 1, 16);
      v += __shfl_xor(v, 2, 16);
      v += __shfl_xor(v, 4, 16);
      v += __shfl_xor(v, 8, 16);
      if ((lane & 15) == 0)
        rowsum[h * NTP + nw + which * 16 + r + 8 * (lane >> 4)] = v;
    }
}

// ---------- stage 5: out = V * softmax(Q^T K)  (fused, streaming over n) ----------
// Block = 4 waves x 32 cols = 128 output columns; double-buffered async q/V chunks.
__global__ __launch_bounds__(128) void attnv4_attn_out(const _Float16* __restrict__ qT,
                                                       const _Float16* __restrict__ kT,
                                                       const _Float16* __restrict__ vC,
                                                       const float* __restrict__ temp,
                                                       const float* __restrict__ rowsum,
                                                       float* __restrict__ outh) {
  __shared__ _Float16 sQ[2][32 * CH];              // 2 x 6 KB
  __shared__ _Float16 sV[2][CH * 32];              // 2 x 6 KB
  __shared__ _Float16 sA[4][32 * 32];              // per-wave A-chunk [m_local][n_local] (8 KB)
  const int h = blockIdx.y;
  const int mblock = blockIdx.x * 128;
  const int wave = threadIdx.x >> 5, lane = threadIdx.x & 31;
  const int mcol = lane & 15, half = lane >> 4;
  const _Float16* q  = qT + (size_t)h * NTP * CH;
  const _Float16* kk = kT + (size_t)h * NTP * CH;
  const _Float16* v  = vC + (size_t)h * CH * NTP;
  const float* rs = rowsum + h * NTP;
  const float th = temp[h];
  const int m0w = mblock + wave * 32;

  v16h bK[2][3];                                   // this wave's 32 key columns (hoisted)
#pragma unroll
  for (int mi = 0; mi < 2; ++mi)
#pragma unroll
    for (int kc = 0; kc < 3; ++kc) bK[mi][kc] = load_b_f16(kk, CH, m0w + mi * 16, kc * 32);

  v8f acc[6][2];
#pragma unroll
  for (int ct = 0; ct < 6; ++ct)
#pragma unroll
    for (int mi = 0; mi < 2; ++mi) acc[ct][mi] = (v8f){};

  _Float16* sAw = &sA[wave][0];

  auto stageQV = [&](int n0, int buf) {            // 6 async instructions per wave
    const _Float16* qsrc = q + (size_t)n0 * CH;    // contiguous 6144 B
    for (int g = threadIdx.x; g < 384; g += 128) cp16B(&sQ[buf][g * 8], qsrc + g * 8);
    for (int g = threadIdx.x; g < 384; g += 128) {
      const int r = g >> 2, o = (g & 3) * 8;
      cp16B(&sV[buf][r * 32 + o], v + (size_t)r * NTP + n0 + o);
    }
  };

  const int NIT = NTP / 32;                        // 140
  stageQV(0, 0);
  for (int it = 0; it < NIT; ++it) {
    const int cur = it & 1, nxt = cur ^ 1;
    const int nnext = (it + 1 < NIT) ? (it + 1) * 32 : it * 32;  // clamp keeps wait const
    stageQV(nnext, nxt);
    wait_async<6>();                               // chunk `it` has landed
    __syncthreads();
    const int n0 = it * 32;

    // S chunk (32 n x 32 m for this wave) -> exp/normalize -> sA
#pragma unroll
    for (int which = 0; which < 2; ++which) {
#pragma unroll
      for (int mi = 0; mi < 2; ++mi) {
        v8f s = {};
#pragma unroll
        for (int kc = 0; kc < 3; ++kc)
          s = wmma16(load_a_f16(sQ[cur], CH, which * 16, kc * 32, CH), bK[mi][kc], s);
#pragma unroll
        for (int r = 0; r < 8; ++r) {
          const int nl = which * 16 + r + 8 * half;
          const float aval = __expf(s[r] * th) / rs[n0 + nl];  // padded-n rows meet V==0
          sAw[(mi * 16 + mcol) * 32 + nl] = (_Float16)aval;
        }
      }
    }
    // per-wave LDS ordering guarantees the B-fragment reads below see the writes
    v16h bf[2];
#pragma unroll
    for (int mi = 0; mi < 2; ++mi)
#pragma unroll
      for (int e = 0; e < 16; ++e) bf[mi][e] = sAw[(mi * 16 + mcol) * 32 + half * 16 + e];
#pragma unroll
    for (int ct = 0; ct < 6; ++ct) {
      v16h aV = load_a_f16(sV[cur], 32, ct * 16, 0, 32);
#pragma unroll
      for (int mi = 0; mi < 2; ++mi) acc[ct][mi] = wmma16(aV, bf[mi], acc[ct][mi]);
    }
    __syncthreads();                               // everyone done with buf `cur`
  }
  float* oh = outh + (size_t)h * CH * NTP;
#pragma unroll
  for (int ct = 0; ct < 6; ++ct)
#pragma unroll
    for (int mi = 0; mi < 2; ++mi) store_d(oh, NTP, ct * 16, m0w + mi * 16, acc[ct][mi]);
}

// ---------- stage 6: fold (non-overlapping -> pure gather) into yT (65536x64 f16) ----------
__global__ void attnv4_fold_yt(const float* __restrict__ outh, _Float16* __restrict__ yT) {
  const int p = blockIdx.x * blockDim.x + threadIdx.x;
  const int y = p >> 8, x = p & 255;
  const int py = y + 4, px = x + 4;
  const int ty = py >> 2, kh = py & 3, tx = px >> 2, kw = px & 3;
  const int t = ty * OHW + tx;
#pragma unroll
  for (int c = 0; c < 64; ++c) {
    float vv = 0.f;
    if (c < CDIM) {
      const int h = c / 6, cp = c % 6;
      const int j = cp * 16 + kh * 4 + kw;
      vv = outh[(size_t)(h * CH + j) * NTP + t];
    }
    yT[(size_t)p * 64 + c] = (_Float16)vv;
  }
}

// ---------- stage 7: out = w_proj(48x48) * y -> (48x65536) f32 ----------
__global__ __launch_bounds__(256) void attnv4_gemm_proj(const float* __restrict__ wp,
                                                        const _Float16* __restrict__ yT,
                                                        float* __restrict__ out) {
  const int wid = blockIdx.x * 8 + (threadIdx.x >> 5);
  const int mt = wid % 3, nt = wid / 3;
  v8f c = {};
#pragma unroll
  for (int k0 = 0; k0 < 64; k0 += 32) {
    v16h a = load_a_f32(wp, CDIM, mt * 16, k0, CDIM);
    v16h b = load_b_f16(yT, 64, nt * 16, k0);
    c = wmma16(a, b, c);
  }
  store_d(out, NPIX, mt * 16, nt * 16, c);
}

extern "C" void kernel_launch(void* const* d_in, const int* in_sizes, int n_in,
                              void* d_out, int out_size, void* d_ws, size_t ws_size,
                              hipStream_t stream) {
  const float* x      = (const float*)d_in[0];
  const float* w_qkv  = (const float*)d_in[1];
  const float* w_dw   = (const float*)d_in[2];
  const float* temp   = (const float*)d_in[3];
  const float* w_proj = (const float*)d_in[4];
  float* out = (float*)d_out;

  char* ws = (char*)d_ws;
  size_t off = 0;
  auto alloc = [&](size_t bytes) -> void* {
    void* p = ws + off;
    off = (off + bytes + 255) & ~(size_t)255;
    return p;
  };
  _Float16* xT   = (_Float16*)alloc((size_t)NPIX * 64 * 2);
  float*    qkvr = (float*)   alloc((size_t)CO * NPIX * 4);
  float*    qkvd = (float*)   alloc((size_t)CO * NPIX * 4);
  _Float16* qT   = (_Float16*)alloc((size_t)NH * NTP * CH * 2);
  _Float16* kT   = (_Float16*)alloc((size_t)NH * NTP * CH * 2);
  _Float16* vC   = (_Float16*)alloc((size_t)NH * CH * NTP * 2);
  float*    rsum = (float*)   alloc((size_t)NH * NTP * 4);
  float*    outh = (float*)   alloc((size_t)NH * CH * NTP * 4);
  _Float16* yT   = (_Float16*)alloc((size_t)NPIX * 64 * 2);

  attnv4_x_to_xt   <<<NPIX / 256, 256, 0, stream>>>(x, xT);
  attnv4_gemm_qkv  <<<(9 * (NPIX / 16)) / 8, 256, 0, stream>>>(w_qkv, xT, qkvr);
  attnv4_dwconv    <<<(CO * NPIX) / 256, 256, 0, stream>>>(qkvr, w_dw, qkvd);
  attnv4_make_heads<<<(NH * NTP) / 256, 256, 0, stream>>>(qkvd, qT, kT, vC);
  attnv4_rowsum    <<<dim3(NTP / 128, NH), 128, 0, stream>>>(qT, kT, temp, rsum);
  attnv4_attn_out  <<<dim3(NTP / 128, NH), 128, 0, stream>>>(qT, kT, vC, temp, rsum, outh);
  attnv4_fold_yt   <<<NPIX / 256, 256, 0, stream>>>(outh, yT);
  attnv4_gemm_proj <<<(3 * (NPIX / 16)) / 8, 256, 0, stream>>>(w_proj, yT, out);
}
